// VGG16_59176059404536
// MI455X (gfx1250) — matrix operand order repaired and checked by reading, not compiled, not run on hardware
//
#include <hip/hip_runtime.h>
#include <hip/hip_bf16.h>

typedef __attribute__((ext_vector_type(16))) _Float16 v16h;
typedef __attribute__((ext_vector_type(8)))  _Float16 v8h;
typedef __attribute__((ext_vector_type(8)))  float    v8f;
typedef unsigned int u32x4 __attribute__((ext_vector_type(4)));
typedef unsigned int u32x8 __attribute__((ext_vector_type(8)));

// ---------------------------------------------------------------------------
// f32 -> f16 convert
// ---------------------------------------------------------------------------
__global__ void f32_to_f16_kernel(const float* __restrict__ in,
                                  _Float16* __restrict__ out, int n) {
  int i = blockIdx.x * blockDim.x + threadIdx.x;
  if (i < n) out[i] = (_Float16)in[i];
}

// ---------------------------------------------------------------------------
// Weight transform: f32 OIHW [Cout][K] -> f16 K-major [Kpad][Npad], zero pad.
// Npad is a multiple of 64 so a 4-subtile N group never needs bounds checks.
// ---------------------------------------------------------------------------
__global__ void wtrans_kernel(const float* __restrict__ w,
                              _Float16* __restrict__ wt,
                              int Cout, int K, int Npad, int Kpad) {
  int i = blockIdx.x * blockDim.x + threadIdx.x;
  int total = Kpad * Npad;
  if (i >= total) return;
  int co = i % Npad;
  int k  = i / Npad;
  _Float16 v = (_Float16)0.f;
  if (co < Cout && k < K) v = (_Float16)w[(size_t)co * K + k];
  wt[i] = v;
}

// ---------------------------------------------------------------------------
// Implicit-GEMM conv, WMMA f32_16x16x32_f16, templated on kernel size so the
// per-element K decode uses compile-time divisors (magic multiplies).
//   Block = 256 threads = 8 waves. blockIdx.y*8+wv = M tile (16 pixels),
//   blockIdx.x = N group of 64 couts (4 subtiles of 16).
//   Each K step: block stages a 32x64 f16 B slab into LDS (one 16B vector
//   store per thread), each wave builds one A fragment and issues 4 WMMAs.
// ---------------------------------------------------------------------------
template <int KSZ>
__global__ void conv_wmma_kernel(const _Float16* __restrict__ in,
                                 const _Float16* __restrict__ wt,
                                 const float* __restrict__ bias,
                                 _Float16* __restrict__ out16,
                                 float* __restrict__ out32,
                                 int Cin, int H, int W,
                                 int Cout, int OH, int OW,
                                 int pad, int dil,
                                 int K, int Kpad, int Npad,
                                 int CoutTot, int coff, int relu, int M) {
  constexpr int KSZ2 = KSZ * KSZ;
  __shared__ __align__(128) _Float16 lb[32 * 64];   // 4KB B slab

  int tid  = threadIdx.x;
  int lane = tid & 31;
  int wv   = tid >> 5;
  int m0   = (blockIdx.y * 8 + wv) * 16;            // M tile base
  int n0g  = blockIdx.x * 64;                       // N group base

  unsigned ohow = (unsigned)(OH * OW);
  int m = m0 + (lane & 15);
  bool mvalid = (m < M);
  unsigned mc = mvalid ? (unsigned)m : 0u;
  int b   = (int)(mc / ohow);
  int pix = (int)(mc - (unsigned)b * ohow);
  int oy  = pix / OW;
  int ox  = pix - oy * OW;
  int kbase = (lane < 16) ? 0 : 8;

  // cooperative stage coordinates
  int srow = tid >> 3;          // 0..31 (K row within slab)
  int scol = (tid & 7) * 8;     // 0..56 (col within 64, 8 halves = 16B)
  const _Float16* wsrc = wt + n0g + scol;

  v8f acc[4] = {};
  for (int k0 = 0; k0 < K; k0 += 32) {
    // ---- stage B slab ----
    *(v8h*)(lb + srow * 64 + scol) =
        *(const v8h*)(wsrc + (size_t)(k0 + srow) * Npad);
    if (k0 + 32 < Kpad)
      __builtin_prefetch(wsrc + (size_t)(k0 + 32 + srow) * Npad, 0, 1);
    __syncthreads();

    // ---- A fragment (implicit im2col gather) ----
    v16h a;
#pragma unroll
    for (int v = 0; v < 8; ++v) {
      int kp = ((v < 4) ? (2 * v) : (16 + 2 * (v - 4))) + kbase;
#pragma unroll
      for (int h = 0; h < 2; ++h) {
        int k = k0 + kp + h;
        _Float16 av = (_Float16)0.f;
        if (mvalid && k < K) {
          int ci = k / KSZ2;                 // compile-time divisor
          int r  = k - ci * KSZ2;
          int kh = r / KSZ;
          int kw = r - kh * KSZ;
          int iy = oy - pad + kh * dil;
          int ix = ox - pad + kw * dil;
          if ((unsigned)iy < (unsigned)H && (unsigned)ix < (unsigned)W)
            av = in[(((size_t)b * Cin + ci) * H + iy) * W + ix];
        }
        a[2 * v + h] = av;
      }
    }

    // ---- 4 WMMAs sharing one A fragment ----
#pragma unroll
    for (int s = 0; s < 4; ++s) {
      v16h bf;
#pragma unroll
      for (int e = 0; e < 16; ++e)
        bf[e] = lb[(((e < 8) ? e : e + 8) + kbase) * 64 + s * 16 + (lane & 15)];
      acc[s] = __builtin_amdgcn_wmma_f32_16x16x32_f16(false, a, false, bf,
                                                      (short)0, acc[s],
                                                      false, false);
    }
    __syncthreads();
  }

  // ---- store ----
  int rbase = (lane < 16) ? 0 : 8;
  int bb[8], yy[8], xx[8];
#pragma unroll
  for (int r = 0; r < 8; ++r) {
    unsigned mr = (unsigned)(m0 + rbase + r);
    unsigned mrc = (mr < (unsigned)M) ? mr : 0u;
    bb[r] = (int)(mrc / ohow);
    int pp = (int)(mrc - (unsigned)bb[r] * ohow);
    yy[r] = pp / OW;
    xx[r] = pp - yy[r] * OW;
  }
#pragma unroll
  for (int s = 0; s < 4; ++s) {
    int ncol = n0g + s * 16 + (lane & 15);
    bool nvalid = ncol < Cout;
    float bz = nvalid ? bias[ncol] : 0.f;
#pragma unroll
    for (int r = 0; r < 8; ++r) {
      int mr = m0 + rbase + r;
      if (nvalid && mr < M) {
        float vs = acc[s][r] + bz;
        if (relu) vs = fmaxf(vs, 0.f);
        size_t oidx =
            (((size_t)bb[r] * CoutTot + coff + ncol) * OH + yy[r]) * OW + xx[r];
        if (out16) out16[oidx] = (_Float16)vs;
        if (out32) out32[oidx] = vs;
      }
    }
  }
}

// ---------------------------------------------------------------------------
// Self-correlation GEMM: corr[b,n,m] = sum_c xn[b,c,n]*xn[b,c,m].
// grid: x = col group (N/64), y = row-tile group (N/16/8), z = batch.
// Each wave: one 16-row tile x 4 col subtiles.
// B slab (32 K-rows x 64 cols f16 = 32 x 128B, row pitch N*2 bytes) is
// staged into LDS by the Tensor Data Mover: wave 0 builds a 2D D# descriptor
// in SGPRs and issues tensor_load_to_lds, then waits TENSORcnt==0.
// N multiple of 64, C multiple of 32 -> no guards needed.
// ---------------------------------------------------------------------------
__global__ void corr_wmma_kernel(const _Float16* __restrict__ xn,
                                 float* __restrict__ corr,
                                 int C, int N) {
  __shared__ __align__(128) _Float16 lb[32 * 64];

  int tid  = threadIdx.x;
  int lane = tid & 31;
  int wv   = tid >> 5;
  int rt   = blockIdx.y * 8 + wv;     // row tile index
  int m0g  = blockIdx.x * 64;         // col group base
  int b    = blockIdx.z;

  const _Float16* base = xn + (size_t)b * C * N;
  int nrow  = rt * 16 + (lane & 15);
  int kbase = (lane < 16) ? 0 : 8;

  // TDM descriptor invariants (units of data_size = 8 bytes)
  unsigned lds_addr = (unsigned)(unsigned long long)(size_t)&lb[0]; // addr[31:0] = LDS offset
  unsigned long long gbase = (unsigned long long)(size_t)(base + m0g);
  unsigned pitch8 = (unsigned)(N * 2 / 8);   // row pitch in 8B units (=1024)

  v8f acc[4] = {};
  for (int k0 = 0; k0 < C; k0 += 32) {
    __syncthreads();   // previous slab fully consumed
    if (wv == 0) {
      unsigned long long gaddr = gbase + (unsigned long long)k0 * (unsigned)(N * 2);
      u32x4 g0;
      g0[0] = 1u;                                   // count=1, user mode
      g0[1] = lds_addr;                             // lds_addr
      g0[2] = (unsigned)gaddr;                      // global_addr[31:0]
      g0[3] = (unsigned)((gaddr >> 32) & 0x1FFFFFFull) | (2u << 30); // addr[56:32], type=2
      u32x8 g1;
      g1[0] = (3u << 16);                           // wg_mask=0, data_size=8B
      g1[1] = (pitch8 & 0xFFFFu) << 16;             // tensor_dim0[15:0]
      g1[2] = (pitch8 >> 16) | ((unsigned)C << 16); // tensor_dim0[31:16] | tensor_dim1[15:0]
      g1[3] = ((unsigned)C >> 16) | (16u << 16);    // tensor_dim1[31:16] | tile_dim0=16
      g1[4] = 32u;                                  // tile_dim1=32, tile_dim2=0
      g1[5] = pitch8;                               // tensor_dim0_stride[31:0]
      g1[6] = 0u;                                   // stride0[47:32] | stride1[15:0]
      g1[7] = 0u;
      asm volatile("tensor_load_to_lds %0, %1" :: "s"(g0), "s"(g1) : "memory");
      __builtin_amdgcn_s_wait_tensorcnt(0);
    }
    __syncthreads();   // slab visible to all waves

    v16h a;
#pragma unroll
    for (int v = 0; v < 8; ++v) {
      int kp = ((v < 4) ? (2 * v) : (16 + 2 * (v - 4))) + kbase;
      a[2 * v]     = base[(size_t)(k0 + kp) * N + nrow];
      a[2 * v + 1] = base[(size_t)(k0 + kp + 1) * N + nrow];
    }

#pragma unroll
    for (int s = 0; s < 4; ++s) {
      v16h bf;
#pragma unroll
      for (int e = 0; e < 16; ++e)
        bf[e] = lb[(((e < 8) ? e : e + 8) + kbase) * 64 + s * 16 + (lane & 15)];
      acc[s] = __builtin_amdgcn_wmma_f32_16x16x32_f16(false, a, false, bf,
                                                      (short)0, acc[s],
                                                      false, false);
    }
  }

  int rbase = (lane < 16) ? 0 : 8;
  float* cb = corr + (size_t)b * N * N;
#pragma unroll
  for (int s = 0; s < 4; ++s) {
    int mcol = m0g + s * 16 + (lane & 15);
#pragma unroll
    for (int r = 0; r < 8; ++r)
      cb[(size_t)(rt * 16 + rbase + r) * N + mcol] = acc[s][r];
  }
}

// ---------------------------------------------------------------------------
// Top-k(129) per corr row; drop self-sim (rank 0); write ranks 1..128 as
// channels. One 256-thread block per row; row cached in LDS.
// ---------------------------------------------------------------------------
__global__ void topk_kernel(const float* __restrict__ corr,
                            _Float16* __restrict__ feat, int N) {
  __shared__ float row[4096];
  __shared__ float rv[256];
  __shared__ int   ri[256];
  int tid = threadIdx.x;
  int rid = blockIdx.x;          // b*N + n
  int b = rid / N;
  int n = rid - b * N;
  const float* src = corr + (size_t)rid * N;
  for (int i = tid; i < N; i += 256) row[i] = src[i];
  __syncthreads();
  for (int it = 0; it < 129; ++it) {
    float bv = -3.0e38f; int bi = 0;
    for (int i = tid; i < N; i += 256) {
      float v = row[i];
      if (v > bv) { bv = v; bi = i; }
    }
    rv[tid] = bv; ri[tid] = bi;
    __syncthreads();
    for (int s = 128; s > 0; s >>= 1) {
      if (tid < s && rv[tid + s] > rv[tid]) { rv[tid] = rv[tid + s]; ri[tid] = ri[tid + s]; }
      __syncthreads();
    }
    if (tid == 0) {
      if (it > 0)
        feat[((size_t)b * 128 + (it - 1)) * (size_t)N + n] = (_Float16)rv[0];
      row[ri[0]] = -3.0e38f;
    }
    __syncthreads();
  }
}

// ---------------------------------------------------------------------------
// 2x2 stride-2 maxpool (f16)
// ---------------------------------------------------------------------------
__global__ void maxpool2_kernel(const _Float16* __restrict__ in,
                                _Float16* __restrict__ out,
                                int Bn, int C, int H, int W) {
  int OH = H >> 1, OW = W >> 1;
  size_t total = (size_t)Bn * C * OH * OW;
  size_t idx = (size_t)blockIdx.x * blockDim.x + threadIdx.x;
  if (idx >= total) return;
  int ox = (int)(idx % OW); size_t t = idx / OW;
  int oy = (int)(t % OH); t /= OH;
  int c  = (int)(t % C);  int b = (int)(t / C);
  const _Float16* p = in + (((size_t)b * C + c) * H + oy * 2) * W + ox * 2;
  float v0 = (float)p[0], v1 = (float)p[1], v2 = (float)p[W], v3 = (float)p[W + 1];
  out[idx] = (_Float16)fmaxf(fmaxf(v0, v1), fmaxf(v2, v3));
}

// ---------------------------------------------------------------------------
// Bilinear resize, align_corners=True (f16 in/out, channel-offset dest)
// ---------------------------------------------------------------------------
__global__ void resize_ac_f16_kernel(const _Float16* __restrict__ in,
                                     _Float16* __restrict__ out,
                                     int Bn, int C, int H, int W,
                                     int OH, int OW, int CoutTot, int coff) {
  size_t total = (size_t)Bn * C * OH * OW;
  size_t idx = (size_t)blockIdx.x * blockDim.x + threadIdx.x;
  if (idx >= total) return;
  int ox = (int)(idx % OW); size_t t = idx / OW;
  int oy = (int)(t % OH); t /= OH;
  int c  = (int)(t % C);  int b = (int)(t / C);
  float fy = (OH > 1) ? (float)oy * (float)(H - 1) / (float)(OH - 1) : 0.f;
  float fx = (OW > 1) ? (float)ox * (float)(W - 1) / (float)(OW - 1) : 0.f;
  int y0 = (int)floorf(fy); int y1 = min(y0 + 1, H - 1); float wy = fy - (float)y0;
  int x0 = (int)floorf(fx); int x1 = min(x0 + 1, W - 1); float wx = fx - (float)x0;
  const _Float16* p = in + ((size_t)b * C + c) * H * W;
  float v00 = (float)p[(size_t)y0 * W + x0], v01 = (float)p[(size_t)y0 * W + x1];
  float v10 = (float)p[(size_t)y1 * W + x0], v11 = (float)p[(size_t)y1 * W + x1];
  float r0 = v00 + (v01 - v00) * wx;
  float r1 = v10 + (v11 - v10) * wx;
  out[(((size_t)b * CoutTot + coff + c) * OH + oy) * OW + ox] =
      (_Float16)(r0 + (r1 - r0) * wy);
}

// Final resize: f32 [B,1,H,W] -> f32 [B,OH,OW]
__global__ void resize_ac_f32_kernel(const float* __restrict__ in,
                                     float* __restrict__ out,
                                     int Bn, int H, int W, int OH, int OW) {
  size_t total = (size_t)Bn * OH * OW;
  size_t idx = (size_t)blockIdx.x * blockDim.x + threadIdx.x;
  if (idx >= total) return;
  int ox = (int)(idx % OW); size_t t = idx / OW;
  int oy = (int)(t % OH); int b = (int)(t / OH);
  float fy = (OH > 1) ? (float)oy * (float)(H - 1) / (float)(OH - 1) : 0.f;
  float fx = (OW > 1) ? (float)ox * (float)(W - 1) / (float)(OW - 1) : 0.f;
  int y0 = (int)floorf(fy); int y1 = min(y0 + 1, H - 1); float wy = fy - (float)y0;
  int x0 = (int)floorf(fx); int x1 = min(x0 + 1, W - 1); float wx = fx - (float)x0;
  const float* p = in + (size_t)b * H * W;
  float v00 = p[(size_t)y0 * W + x0], v01 = p[(size_t)y0 * W + x1];
  float v10 = p[(size_t)y1 * W + x0], v11 = p[(size_t)y1 * W + x1];
  float r0 = v00 + (v01 - v00) * wx;
  float r1 = v10 + (v11 - v10) * wx;
  out[idx] = r0 + (r1 - r0) * wy;
}

// ---------------------------------------------------------------------------
// x_seg column normalize: [B,Hh,Ww] normalized over axis=1 (rows), f16 out.
// ---------------------------------------------------------------------------
__global__ void xseg_norm_kernel(const float* __restrict__ xs,
                                 _Float16* __restrict__ o,
                                 int Bn, int Hh, int Ww) {
  int idx = blockIdx.x * blockDim.x + threadIdx.x;
  if (idx >= Bn * Ww) return;
  int b = idx / Ww, w = idx - (idx / Ww) * Ww;
  float s = 0.f;
  for (int h = 0; h < Hh; ++h) {
    float v = xs[((size_t)b * Hh + h) * Ww + w];
    s += v * v;
  }
  float inv = rsqrtf(fmaxf(s, 1e-24f));
  for (int h = 0; h < Hh; ++h) {
    float v = xs[((size_t)b * Hh + h) * Ww + w];
    o[((size_t)b * Hh + h) * Ww + w] = (_Float16)(v * inv);
  }
}

// ---------------------------------------------------------------------------
// Per-pixel channel L2 norm over C channels (NCHW, spatial flattened to N)
// ---------------------------------------------------------------------------
__global__ void chan_l2norm_kernel(const _Float16* __restrict__ in,
                                   _Float16* __restrict__ out,
                                   int Bn, int C, int N) {
  int idx = blockIdx.x * blockDim.x + threadIdx.x;
  if (idx >= Bn * N) return;
  int b = idx / N;
  int n = idx - b * N;
  const _Float16* p = in + (size_t)b * C * N + n;
  float s = 0.f;
  for (int c = 0; c < C; ++c) { float v = (float)p[(size_t)c * N]; s += v * v; }
  float inv = rsqrtf(fmaxf(s, 1e-24f));
  _Float16* q = out + (size_t)b * C * N + n;
  for (int c = 0; c < C; ++c)
    q[(size_t)c * N] = (_Float16)((float)p[(size_t)c * N] * inv);
}

// ---------------------------------------------------------------------------
// Host side
// ---------------------------------------------------------------------------
static inline int cdivi(long long a, long long b) { return (int)((a + b - 1) / b); }

extern "C" void kernel_launch(void* const* d_in, const int* in_sizes, int n_in,
                              void* d_out, int out_size, void* d_ws, size_t ws_size,
                              hipStream_t stream) {
  (void)n_in; (void)out_size; (void)ws_size;
  const int B = 2;

  // ---- workspace layout (256B aligned) ----
  char* base = (char*)d_ws;
  size_t off = 0;
  auto alloc = [&](size_t bytes) -> void* {
    void* p = base + off;
    off += (bytes + 255) & ~(size_t)255;
    return p;
  };
  _Float16* wt   = (_Float16*)alloc(4718592);    // max Kpad*Npad*2 (4608*512)
  _Float16* actA = (_Float16*)alloc(16777216);   // 8,388,608 f16
  _Float16* actB = (_Float16*)alloc(16777216);
  _Float16* s3   = (_Float16*)alloc(4194304);    // 2*256*64*64
  _Float16* s4   = (_Float16*)alloc(2097152);    // 2*512*32*32
  _Float16* s5   = (_Float16*)alloc(524288);     // 2*512*16*16
  _Float16* dcat = (_Float16*)alloc(524288);     // 2*512*16*16
  _Float16* ctmp = (_Float16*)alloc(4194304);    // up to 2*256*64*64
  _Float16* scat = (_Float16*)alloc(20971520);   // 2*1280*64*64
  _Float16* xn   = (_Float16*)alloc(20971520);
  float*    corr = (float*)  alloc(134217728);   // 2*4096*4096 f32
  _Float16* feat = (_Float16*)alloc(2097152);    // 2*128*64*64
  _Float16* xsn  = (_Float16*)alloc(16384);      // 2*1*64*64
  float*    q5   = (float*)  alloc(32768);       // 2*1*56*56

  // ---- conv helper: weight transform + WMMA conv ----
  auto conv = [&](int wi, const _Float16* src, _Float16* o16, float* o32,
                  int Cin, int H, int W, int Cout,
                  int ksz, int pad, int dil, int CoutTot, int coff, int relu) {
    const float* wptr = (const float*)d_in[wi];
    const float* bptr = (const float*)d_in[wi + 1];
    int K = Cin * ksz * ksz;
    int Kpad = (K + 31) & ~31;
    int Npad = (Cout + 63) & ~63;     // multiple of 64 for 4-subtile groups
    int nwt = Kpad * Npad;
    wtrans_kernel<<<cdivi(nwt, 256), 256, 0, stream>>>(wptr, wt, Cout, K, Npad, Kpad);
    int OH = H + 2 * pad - dil * (ksz - 1);
    int OW = W + 2 * pad - dil * (ksz - 1);
    int M = B * OH * OW;
    int Mt = (M + 15) / 16;
    dim3 grid(Npad / 64, cdivi(Mt, 8));
    if (ksz == 3) {
      conv_wmma_kernel<3><<<grid, 256, 0, stream>>>(
          src, wt, bptr, o16, o32, Cin, H, W, Cout, OH, OW, pad, dil,
          K, Kpad, Npad, CoutTot, coff, relu, M);
    } else {
      conv_wmma_kernel<1><<<grid, 256, 0, stream>>>(
          src, wt, bptr, o16, o32, Cin, H, W, Cout, OH, OW, pad, dil,
          K, Kpad, Npad, CoutTot, coff, relu, M);
    }
  };
  auto pool = [&](const _Float16* src, _Float16* dst, int C, int H, int W) {
    size_t total = (size_t)B * C * (H / 2) * (W / 2);
    maxpool2_kernel<<<cdivi((long long)total, 256), 256, 0, stream>>>(src, dst, B, C, H, W);
  };
  auto resize = [&](const _Float16* src, _Float16* dst, int C, int H, int W,
                    int OH, int OW, int CoutTot, int coff) {
    size_t total = (size_t)B * C * OH * OW;
    resize_ac_f16_kernel<<<cdivi((long long)total, 256), 256, 0, stream>>>(
        src, dst, B, C, H, W, OH, OW, CoutTot, coff);
  };

  // ---- input convert ----
  f32_to_f16_kernel<<<cdivi(in_sizes[0], 256), 256, 0, stream>>>(
      (const float*)d_in[0], actA, in_sizes[0]);

  // ---- VGG backbone (weights/biases start at d_in[2]) ----
  conv(2,  actA, actB, nullptr,   3, 256, 256,  64, 3, 1, 1,  64, 0, 1);  // b1c0
  conv(4,  actB, actA, nullptr,  64, 256, 256,  64, 3, 1, 1,  64, 0, 1);  // b1c1
  pool(actA, actB, 64, 256, 256);
  conv(6,  actB, actA, nullptr,  64, 128, 128, 128, 3, 1, 1, 128, 0, 1);  // b2c0
  conv(8,  actA, actB, nullptr, 128, 128, 128, 128, 3, 1, 1, 128, 0, 1);  // b2c1
  pool(actB, actA, 128, 128, 128);
  conv(10, actA, actB, nullptr, 128,  64,  64, 256, 3, 1, 1, 256, 0, 1);  // b3c0
  conv(12, actB, actA, nullptr, 256,  64,  64, 256, 3, 1, 1, 256, 0, 1);  // b3c1
  conv(14, actA, s3,   nullptr, 256,  64,  64, 256, 3, 1, 1, 256, 0, 1);  // b3c2 -> s3
  pool(s3, actA, 256, 64, 64);
  conv(16, actA, actB, nullptr, 256,  32,  32, 512, 3, 1, 1, 512, 0, 1);  // b4c0
  conv(18, actB, actA, nullptr, 512,  32,  32, 512, 3, 1, 1, 512, 0, 1);  // b4c1
  conv(20, actA, s4,   nullptr, 512,  32,  32, 512, 3, 1, 1, 512, 0, 1);  // b4c2 -> s4
  pool(s4, actA, 512, 32, 32);
  conv(22, actA, actB, nullptr, 512,  16,  16, 512, 3, 1, 1, 512, 0, 1);  // b5c0
  conv(24, actB, actA, nullptr, 512,  16,  16, 512, 3, 1, 1, 512, 0, 1);  // b5c1
  conv(26, actA, s5,   nullptr, 512,  16,  16, 512, 3, 1, 1, 512, 0, 1);  // b5c2 -> s5

  // ---- dilated branch -> dcat [B,512,16,16] ----
  conv(28, s5, dcat, nullptr, 512, 16, 16, 128, 3,  2,  2, 512,   0, 1);  // d2
  conv(30, s5, dcat, nullptr, 512, 16, 16, 128, 3,  4,  4, 512, 128, 1);  // d4
  conv(32, s5, dcat, nullptr, 512, 16, 16, 128, 3,  8,  8, 512, 256, 1);  // d8
  conv(34, s5, dcat, nullptr, 512, 16, 16, 128, 3, 16, 16, 512, 384, 1);  // d16

  // ---- 1x1 score convs, resized into scat [B,1280,64,64] ----
  conv(36, dcat, ctmp, nullptr, 512, 16, 16, 256, 1, 0, 1, 256, 0, 1);    // c1
  resize(ctmp, scat, 256, 16, 16, 64, 64, 1280, 0);
  conv(38, s5,   ctmp, nullptr, 512, 16, 16, 256, 1, 0, 1, 256, 0, 1);    // c2
  resize(ctmp, scat, 256, 16, 16, 64, 64, 1280, 256);
  conv(40, s4,   ctmp, nullptr, 512, 32, 32, 256, 1, 0, 1, 256, 0, 1);    // c3
  resize(ctmp, scat, 256, 32, 32, 64, 64, 1280, 512);
  conv(42, s3,   scat, nullptr, 256, 64, 64, 256, 1, 0, 1, 1280, 768, 1); // c4 (64x64)

  // ---- x_seg normalize + cseg conv directly into scat ----
  xseg_norm_kernel<<<cdivi(B * 64, 128), 128, 0, stream>>>(
      (const float*)d_in[1], xsn, B, 64, 64);
  conv(44, xsn, scat, nullptr, 1, 64, 64, 256, 1, 0, 1, 1280, 1024, 1);   // cseg

  // ---- channel L2 norm -> xn; self-correlation GEMM; top-k pooling ----
  const int N = 64 * 64;        // 4096
  const int C = 1280;
  chan_l2norm_kernel<<<cdivi(B * N, 256), 256, 0, stream>>>(scat, xn, B, C, N);
  {
    dim3 grid(N / 64, N / 16 / 8, B);   // 64 x 32 x 2
    corr_wmma_kernel<<<grid, 256, 0, stream>>>(xn, corr, C, N);
  }
  topk_kernel<<<B * N, 256, 0, stream>>>(corr, feat, N);

  // ---- prediction head (pad=0 convs) ----
  conv(46, feat, actA, nullptr, 128, 64, 64, 512, 3, 0, 1, 512, 0, 1);    // 64 -> 62
  conv(48, actA, actB, nullptr, 512, 62, 62, 512, 3, 0, 1, 512, 0, 1);    // 62 -> 60
  conv(50, actB, actA, nullptr, 512, 60, 60, 512, 3, 0, 1, 512, 0, 1);    // 60 -> 58
  conv(52, actA, actB, nullptr, 512, 58, 58,  64, 3, 0, 1,  64, 0, 1);    // 58 -> 56
  conv(54, actB, nullptr, q5,    64, 56, 56,   1, 1, 0, 1,   1, 0, 0);    // 1x1, f32

  // ---- final align-corners resize to [B,256,256] ----
  {
    size_t total = (size_t)B * 256 * 256;
    resize_ac_f32_kernel<<<cdivi((long long)total, 256), 256, 0, stream>>>(
        q5, (float*)d_out, B, 56, 56, 256, 256);
  }
}
